// GCN_46866683134643
// MI455X (gfx1250) — matrix operand order repaired and checked by reading, not compiled, not run on hardware
//
#include <hip/hip_runtime.h>
#include <hip/hip_bf16.h>

typedef __attribute__((ext_vector_type(16))) __bf16 v16bf;
typedef __attribute__((ext_vector_type(8)))  __bf16 v8bf;
typedef __attribute__((ext_vector_type(8)))  float  v8f;

// ---------------- degree / norm precompute ----------------

__global__ void gcn_deg_init(float* __restrict__ deg, int n) {
    int i = blockIdx.x * blockDim.x + threadIdx.x;
    if (i < n) deg[i] = 1.0f;                      // self-loop weight
}

__global__ void gcn_deg_edges(float* __restrict__ deg, const int* __restrict__ dst,
                              const float* __restrict__ w, int E) {
    int e = blockIdx.x * blockDim.x + threadIdx.x;
    if (e < E) atomicAdd(&deg[dst[e]], w[e]);
}

__global__ void gcn_dinv(float* __restrict__ deg, int n) {
    int i = blockIdx.x * blockDim.x + threadIdx.x;
    if (i < n) {
        float d = deg[i];
        deg[i] = (d > 0.0f) ? rsqrtf(d) : 0.0f;    // in-place: deg -> dinv
    }
}

__global__ void gcn_norm(const int* __restrict__ src, const int* __restrict__ dst,
                         const float* __restrict__ w, const float* __restrict__ dinv,
                         float* __restrict__ norm, int E) {
    int e = blockIdx.x * blockDim.x + threadIdx.x;
    if (e < E) norm[e] = dinv[src[e]] * w[e] * dinv[dst[e]];
}

// ---------------- casts ----------------

__global__ void gcn_cast_bf16(const float* __restrict__ in, __bf16* __restrict__ out,
                              long long n, int do_relu) {
    long long i = blockIdx.x * (long long)blockDim.x + threadIdx.x;
    if (i < n) {
        float v = in[i];
        if (do_relu) v = fmaxf(v, 0.0f);
        out[i] = (__bf16)v;
    }
}

// W [K][N] f32 row-major  ->  BT [N][K] bf16 (so B fragment loads are contiguous in K)
__global__ void gcn_w_transpose_bf16(const float* __restrict__ W, __bf16* __restrict__ BT,
                                     int K, int N) {
    int i = blockIdx.x * blockDim.x + threadIdx.x;
    if (i < K * N) {
        int k = i / N, n = i % N;
        BT[(size_t)n * K + k] = (__bf16)W[i];
    }
}

// ---------------- WMMA GEMM: C[M][N] = A[M][K] (bf16) x BT[N][K]^T (bf16), f32 accum ----
// grid.x = M/16 tiles (M multiple of 16), blockDim = (32, N/16): one wave per 16x16 tile.

__global__ void __launch_bounds__(256)
gcn_gemm_bf16_wmma(const __bf16* __restrict__ A, const __bf16* __restrict__ BT,
                   float* __restrict__ C, int N, int K) {
    const int lane = threadIdx.x & 31;
    const int tn   = threadIdx.y;
    const int tm   = blockIdx.x;
    const int r15  = lane & 15;
    const int half = lane >> 4;

    const __bf16* arow = A  + (size_t)(tm * 16 + r15) * K;   // A row for this lane
    const __bf16* bcol = BT + (size_t)(tn * 16 + r15) * K;   // B column for this lane

    v8f acc = {};
    for (int k = 0; k < K; k += 32) {
        const int ka = k + half * 8;
        // A fragment: lanes 0-15 K=[k,k+8)u[k+16,k+24); lanes 16-31 K=[k+8,k+16)u[k+24,k+32)
        v8bf alo = *(const v8bf*)(arow + ka);
        v8bf ahi = *(const v8bf*)(arow + ka + 16);
        v16bf a = __builtin_shufflevector(alo, ahi,
                                          0, 1, 2, 3, 4, 5, 6, 7,
                                          8, 9, 10, 11, 12, 13, 14, 15);
        // B fragment: same K layout as A: lanes 0-15 K=[k,k+8)u[k+16,k+24); lanes 16-31 K=[k+8,k+16)u[k+24,k+32)
        v8bf blo = *(const v8bf*)(bcol + ka);
        v8bf bhi = *(const v8bf*)(bcol + ka + 16);
        v16bf b = __builtin_shufflevector(blo, bhi,
                                          0, 1, 2, 3, 4, 5, 6, 7,
                                          8, 9, 10, 11, 12, 13, 14, 15);
        acc = __builtin_amdgcn_wmma_f32_16x16x32_bf16(
                  false, a, false, b, (short)0, acc, false, false);
    }
    // C/D layout: VGPR r -> row (half*8 + r), col = lane&15
    float* cp = C + (size_t)(tm * 16 + half * 8) * N + tn * 16 + r15;
#pragma unroll
    for (int r = 0; r < 8; ++r) cp[(size_t)r * N] = acc[r];
}

// ---------------- aggregation ----------------

// out[i][f] = t[i][f] * dinv[i]^2 + bias[f]   (self-loop + bias; full init of out)
__global__ void gcn_agg_init(const float* __restrict__ t, const float* __restrict__ dinv,
                             const float* __restrict__ bias, float* __restrict__ out,
                             int F, int nN) {
    long long idx = blockIdx.x * (long long)blockDim.x + threadIdx.x;
    long long total = (long long)nN * F;
    if (idx < total) {
        int i = (int)(idx / F);
        int f = (int)(idx % F);
        float di = dinv[i];
        out[idx] = t[idx] * di * di + bias[f];
    }
}

// out[dst][f] += t[src][f] * norm[e], 4 features per thread
__global__ void gcn_agg_edges(const float* __restrict__ t, const int* __restrict__ src,
                              const int* __restrict__ dst, const float* __restrict__ norm,
                              float* __restrict__ out, int F, int E) {
    long long idx = blockIdx.x * (long long)blockDim.x + threadIdx.x;
    int groups = F >> 2;                      // power of two (32/16/8)
    long long total = (long long)E * groups;
    if (idx >= total) return;
    int e = (int)(idx / groups);
    int g = (int)(idx % groups) << 2;
    float nm = norm[e];
    int s = src[e], d = dst[e];
    const float4 v = *(const float4*)(t + (size_t)s * F + g);
    float* o = out + (size_t)d * F + g;
    atomicAdd(o + 0, v.x * nm);
    atomicAdd(o + 1, v.y * nm);
    atomicAdd(o + 2, v.z * nm);
    atomicAdd(o + 3, v.w * nm);
}

// ---------------- host launch ----------------

static inline size_t align256(size_t x) { return (x + 255) & ~(size_t)255; }

extern "C" void kernel_launch(void* const* d_in, const int* in_sizes, int n_in,
                              void* d_out, int out_size, void* d_ws, size_t ws_size,
                              hipStream_t stream) {
    const float* x  = (const float*)d_in[0];
    const int*   ei = (const int*)d_in[1];      // edge_index, [2][E]
    const float* ea = (const float*)d_in[2];
    const float* W1 = (const float*)d_in[3]; const float* b1 = (const float*)d_in[4];
    const float* W2 = (const float*)d_in[5]; const float* b2 = (const float*)d_in[6];
    const float* W3 = (const float*)d_in[7]; const float* b3 = (const float*)d_in[8];

    const int D  = 128;
    const int Nn = in_sizes[0] / D;            // 50000 (multiple of 16)
    const int E  = in_sizes[2];                // 800000
    const int* src = ei;
    const int* dst = ei + E;

    // workspace carve-out
    char* ws = (char*)d_ws;
    size_t off = 0;
    float*  dinv = (float*)(ws + off); off = align256(off + (size_t)Nn * 4);
    float*  norm = (float*)(ws + off); off = align256(off + (size_t)E * 4);
    __bf16* hb   = (__bf16*)(ws + off); off = align256(off + (size_t)Nn * D * 2);
    __bf16* wbt  = (__bf16*)(ws + off); off = align256(off + (size_t)D * D * 2);
    float*  tbuf = (float*)(ws + off); off = align256(off + (size_t)Nn * D * 4);
    float*  agg  = (float*)(ws + off); off = align256(off + (size_t)Nn * D * 4);

    const int B = 256;
    auto nb = [](long long n, int b) { return (unsigned)((n + b - 1) / b); };

    // ---- degree / norm ----
    gcn_deg_init <<<nb(Nn, B), B, 0, stream>>>(dinv, Nn);
    gcn_deg_edges<<<nb(E, B),  B, 0, stream>>>(dinv, dst, ea, E);
    gcn_dinv     <<<nb(Nn, B), B, 0, stream>>>(dinv, Nn);
    gcn_norm     <<<nb(E, B),  B, 0, stream>>>(src, dst, ea, dinv, norm, E);

    // ---- layer 1: 128 -> 128 ----
    gcn_cast_bf16<<<nb((long long)Nn * 128, B), B, 0, stream>>>(x, hb, (long long)Nn * 128, 0);
    gcn_w_transpose_bf16<<<nb(128 * 128, B), B, 0, stream>>>(W1, wbt, 128, 128);
    gcn_gemm_bf16_wmma<<<dim3(Nn / 16), dim3(32, 8), 0, stream>>>(hb, wbt, tbuf, 128, 128);
    gcn_agg_init <<<nb((long long)Nn * 128, B), B, 0, stream>>>(tbuf, dinv, b1, agg, 128, Nn);
    gcn_agg_edges<<<nb((long long)E * 32, B), B, 0, stream>>>(tbuf, src, dst, norm, agg, 128, E);

    // ---- layer 2: 128 -> 64 ----
    gcn_cast_bf16<<<nb((long long)Nn * 128, B), B, 0, stream>>>(agg, hb, (long long)Nn * 128, 1);
    gcn_w_transpose_bf16<<<nb(128 * 64, B), B, 0, stream>>>(W2, wbt, 128, 64);
    gcn_gemm_bf16_wmma<<<dim3(Nn / 16), dim3(32, 4), 0, stream>>>(hb, wbt, tbuf, 64, 128);
    gcn_agg_init <<<nb((long long)Nn * 64, B), B, 0, stream>>>(tbuf, dinv, b2, agg, 64, Nn);
    gcn_agg_edges<<<nb((long long)E * 16, B), B, 0, stream>>>(tbuf, src, dst, norm, agg, 64, E);

    // ---- layer 3: 64 -> 32 (no relu, straight to d_out) ----
    gcn_cast_bf16<<<nb((long long)Nn * 64, B), B, 0, stream>>>(agg, hb, (long long)Nn * 64, 1);
    gcn_w_transpose_bf16<<<nb(64 * 32, B), B, 0, stream>>>(W3, wbt, 64, 32);
    gcn_gemm_bf16_wmma<<<dim3(Nn / 16), dim3(32, 2), 0, stream>>>(hb, wbt, tbuf, 32, 64);
    float* out = (float*)d_out;
    gcn_agg_init <<<nb((long long)Nn * 32, B), B, 0, stream>>>(tbuf, dinv, b3, out, 32, Nn);
    gcn_agg_edges<<<nb((long long)E * 8, B), B, 0, stream>>>(tbuf, src, dst, norm, out, 32, E);
}
